// WSTOG_12326556139683
// MI455X (gfx1250) — matrix operand order repaired and checked by reading, not compiled, not run on hardware
//
#include <hip/hip_runtime.h>
#include <hip/hip_bf16.h>
#include <math.h>

// ---------------- problem constants ----------------
#define D_     4096
#define OBJ_   20
#define M_     1000
#define B_     4
#define Q_     4
#define F_     20
#define ROWB   (OBJ_ * D_)          // 81920 floats per memory / frame row
#define NSTR   1008                 // padded N (63*16): rows of mem*_mp and S stride
#define NT_N   63                   // NSTR / 16
#define RP_ELEMS ((size_t)Q_ * B_ * F_ * OBJ_ * D_)   // 26,214,400

typedef __attribute__((ext_vector_type(16))) _Float16 v16h;
typedef __attribute__((ext_vector_type(8)))  _Float16 v8h;
typedef __attribute__((ext_vector_type(8)))  float    v8f;

// ---------------- workspace layout ----------------
// f16 region first (in halves), then f32 region, then int region.
#define H_MEM1MP ((size_t)0)                          // NSTR*4096 halves (padded rows zeroed)
#define H_MEM2MP (H_MEM1MP + (size_t)NSTR * D_)
#define H_RMAX   (H_MEM2MP + (size_t)NSTR * D_)       // 80*4096
#define H_RHLMP  (H_RMAX + (size_t)80 * D_)           // 32*4096
#define H_TOTAL  (H_RHLMP + (size_t)32 * D_)          // halves (x2 bytes, 16B aligned)

#define FOFF_S1      ((size_t)0)                      // 80*NSTR f32
#define FOFF_S2      (FOFF_S1 + (size_t)80 * NSTR)
#define FOFF_SH      (FOFF_S2 + (size_t)80 * NSTR)    // 16*NSTR
#define FOFF_SL      (FOFF_SH + (size_t)16 * NSTR)
#define FOFF_RSCALE  (FOFF_SL + (size_t)16 * NSTR)    // 1600
#define FOFF_INTS    (FOFF_RSCALE + 1600)
// int region layout
#define IHIGH  0     // 16
#define ILOW   16    // 16
#define IMAP   32    // 32 (rowmap for high/low maxpool)
#define ISIM1  64    // 80
#define ISIM2  144   // 80
#define IHSIM  224   // 16
#define ILSIM  240   // 16

// ------------------------------------------------------------------
// K0: zero the 8 padding rows (1000..1007) of both mem mp banks
// ------------------------------------------------------------------
__global__ __launch_bounds__(256) void zero_pad_kernel(_Float16* __restrict__ mp1,
                                                       _Float16* __restrict__ mp2) {
  _Float16* base = (blockIdx.x < 8) ? (mp1 + (size_t)(M_ + blockIdx.x) * D_)
                                    : (mp2 + (size_t)(M_ + blockIdx.x - 8) * D_);
  unsigned int* p = (unsigned int*)base;          // 2048 dwords per row
  for (int i = threadIdx.x; i < D_ / 2; i += 256) p[i] = 0u;
}

// ------------------------------------------------------------------
// K1: per memory row m: max over OBJ, L2-normalize -> f16 mp[m, 0:4096]
// ------------------------------------------------------------------
__global__ __launch_bounds__(256) void memmp_kernel(const float* __restrict__ mem,
                                                    _Float16* __restrict__ mp) {
  __shared__ float red[256];
  const int m = blockIdx.x;
  const float* base = mem + (size_t)m * ROWB;
  float vals[16];
  float ss = 0.f;
#pragma unroll
  for (int i = 0; i < 16; ++i) {
    const int d = threadIdx.x + i * 256;
    float mx = -INFINITY;
#pragma unroll
    for (int o = 0; o < OBJ_; ++o) mx = fmaxf(mx, base[o * D_ + d]);
    vals[i] = mx;
    ss += mx * mx;
  }
  red[threadIdx.x] = ss;
  __syncthreads();
  for (int s = 128; s > 0; s >>= 1) {
    if (threadIdx.x < (unsigned)s) red[threadIdx.x] += red[threadIdx.x + s];
    __syncthreads();
  }
  const float scale = 1.0f / fmaxf(sqrtf(red[0]), 1e-12f);
  _Float16* out = mp + (size_t)m * D_;
#pragma unroll
  for (int i = 0; i < 16; ++i)
    out[threadIdx.x + i * 256] = (_Float16)(vals[i] * scale);
}

// ------------------------------------------------------------------
// K2: per (b,f,obj) row of input: rscale[row] = 1/max(||row||,eps)
// ------------------------------------------------------------------
__global__ __launch_bounds__(256) void rscale_kernel(const float* __restrict__ x,
                                                     float* __restrict__ scale) {
  __shared__ float red[256];
  const size_t row = blockIdx.x;                 // 0..1599
  const float* p = x + row * D_;
  float ss = 0.f;
  for (int i = threadIdx.x; i < D_; i += 256) { const float v = p[i]; ss += v * v; }
  red[threadIdx.x] = ss;
  __syncthreads();
  for (int s = 128; s > 0; s >>= 1) {
    if (threadIdx.x < (unsigned)s) red[threadIdx.x] += red[threadIdx.x + s];
    __syncthreads();
  }
  if (threadIdx.x == 0) scale[row] = 1.0f / fmaxf(sqrtf(red[0]), 1e-12f);
}

// ------------------------------------------------------------------
// K3/K5: per output row: max over OBJ of normalized R, then l2norm -> f16.
// rowmap==nullptr -> source row = blockIdx.x (all B*F rows for Rmax);
// else source (b*F+f) comes from rowmap[blockIdx.x] (high/low gather).
// ------------------------------------------------------------------
__global__ __launch_bounds__(256) void maxpool_norm_kernel(const float* __restrict__ input,
                                                           const float* __restrict__ rscale,
                                                           const int* __restrict__ rowmap,
                                                           _Float16* __restrict__ out) {
  __shared__ float red[256];
  const int orow = blockIdx.x;
  const int srow = rowmap ? rowmap[orow] : orow;  // b*F + f
  const float* base = input + (size_t)srow * ROWB;
  const float* sc = rscale + srow * OBJ_;
  float vals[16];
  float ss = 0.f;
#pragma unroll
  for (int i = 0; i < 16; ++i) {
    const int d = threadIdx.x + i * 256;
    float mx = -INFINITY;
#pragma unroll
    for (int o = 0; o < OBJ_; ++o) mx = fmaxf(mx, base[o * D_ + d] * sc[o]);
    vals[i] = mx;
    ss += mx * mx;
  }
  red[threadIdx.x] = ss;
  __syncthreads();
  for (int s = 128; s > 0; s >>= 1) {
    if (threadIdx.x < (unsigned)s) red[threadIdx.x] += red[threadIdx.x + s];
    __syncthreads();
  }
  const float scale = 1.0f / fmaxf(sqrtf(red[0]), 1e-12f);
  _Float16* op = out + (size_t)orow * D_;
#pragma unroll
  for (int i = 0; i < 16; ++i)
    op[threadIdx.x + i * 256] = (_Float16)(vals[i] * scale);
}

// ------------------------------------------------------------------
// K4: per (b,q): high = argmax(w), low = rand-th smallest (stable asc)
// ------------------------------------------------------------------
__global__ void select_kernel(const float* __restrict__ w,
                              const int* __restrict__ rand_idx,
                              int* __restrict__ high_idx, int* __restrict__ low_idx) {
  const int t = threadIdx.x;
  if (t >= 16) return;
  const int b = t >> 2, q = t & 3;
  const float* row = w + (b * Q_ + q) * F_;
  int hi = 0; float hv = row[0];
  for (int f = 1; f < F_; ++f) if (row[f] > hv) { hv = row[f]; hi = f; }
  high_idx[t] = hi;  // keyed by t = b*4+q
  int idxs[F_];
  for (int f = 0; f < F_; ++f) idxs[f] = f;
  for (int i = 0; i < F_; ++i)
    for (int j = i + 1; j < F_; ++j) {
      const float vi = row[idxs[i]], vj = row[idxs[j]];
      if (vj < vi || (vj == vi && idxs[j] < idxs[i])) {
        const int tmp = idxs[i]; idxs[i] = idxs[j]; idxs[j] = tmp;
      }
    }
  low_idx[t] = idxs[rand_idx[b * Q_ + q]];
}

// K4b: build rowmap for high/low maxpool rows (r = q*4+b ordering)
__global__ void rowmap_kernel(const int* __restrict__ high_idx,
                              const int* __restrict__ low_idx,
                              int* __restrict__ rowmap) {
  const int r = threadIdx.x;
  if (r >= 32) return;
  const int t = r & 15;           // q*4+b
  const int q = t >> 2, b = t & 3;
  const int f = (r < 16) ? high_idx[b * Q_ + q] : low_idx[b * Q_ + q];
  rowmap[r] = b * F_ + f;
}

// ------------------------------------------------------------------
// K6: WMMA GEMM  S[M,NSTR] = A[M,4096] * Bm[NSTR,4096]^T  (f16 -> f32)
// One wave32 per 16x16 tile; operands pre-packed f16; all tiles
// fully in-bounds (Bm padded+zeroed to NSTR rows).
// A fragment (16-bit A 16x32): lane<16 -> K 0-15,
// lane>=16 -> K 16-31, M = lane%16.
// B fragment (32x16): N = lane%16, lane half selects K 0-15 / 16-31.
// Inner loop: 8x global_load_b128 + 2x v_wmma per 64 K-elements.
// ------------------------------------------------------------------
__global__ __launch_bounds__(32) void wmma_gemm_nt_kernel(const _Float16* __restrict__ A,
                                                          const _Float16* __restrict__ Bm,
                                                          float* __restrict__ S,
                                                          int ntiles_n) {
  const int mt = blockIdx.x / ntiles_n;
  const int nt = blockIdx.x % ntiles_n;
  const int lane = threadIdx.x;
  const int half = lane >> 4;
  const int lmod = lane & 15;

  const _Float16* Arow = A + (size_t)(mt * 16 + lmod) * D_ + half * 16;  // kA folded in
  const _Float16* Brow = Bm + (size_t)(nt * 16 + lmod) * D_ + half * 16; // kB folded in

  v8f c = {};
  for (int kk = 0; kk < D_; kk += 64) {
    __builtin_prefetch(Arow + kk + 256, 0, 0);
    __builtin_prefetch(Brow + kk + 256, 0, 0);
    // K chunk [kk, kk+32)
    const v8h a0 = *(const v8h*)(Arow + kk);
    const v8h a1 = *(const v8h*)(Arow + kk + 8);
    const v16h aA = __builtin_shufflevector(a0, a1, 0, 1, 2, 3, 4, 5, 6, 7,
                                            8, 9, 10, 11, 12, 13, 14, 15);
    const v16h bA = *(const v16h*)(Brow + kk);
    c = __builtin_amdgcn_wmma_f32_16x16x32_f16(false, aA, false, bA,
                                               (short)0, c, false, false);
    // K chunk [kk+32, kk+64)
    const v8h a2 = *(const v8h*)(Arow + kk + 32);
    const v8h a3 = *(const v8h*)(Arow + kk + 40);
    const v16h aB = __builtin_shufflevector(a2, a3, 0, 1, 2, 3, 4, 5, 6, 7,
                                            8, 9, 10, 11, 12, 13, 14, 15);
    const v16h bB = *(const v16h*)(Brow + kk + 32);
    c = __builtin_amdgcn_wmma_f32_16x16x32_f16(false, aB, false, bB,
                                               (short)0, c, false, false);
  }
  // C/D layout: VGPR r -> M = r + half*8, N = lane%16
  const int nIdx = nt * 16 + lmod;
#pragma unroll
  for (int r = 0; r < 8; ++r) {
    const int m = mt * 16 + r + half * 8;
    S[(size_t)m * NSTR + nIdx] = c[r];
  }
}

// ------------------------------------------------------------------
// K7: per-row argmax (first occurrence) over n_cols, stride NSTR
// ------------------------------------------------------------------
__global__ __launch_bounds__(256) void argmax_kernel(const float* __restrict__ S,
                                                     int n_cols, int* __restrict__ out) {
  __shared__ float rv[256];
  __shared__ int   ri[256];
  const float* row = S + (size_t)blockIdx.x * NSTR;
  float bv = -INFINITY; int bi = 0x7fffffff;
  for (int j = threadIdx.x; j < n_cols; j += 256) {
    const float v = row[j];
    if (v > bv) { bv = v; bi = j; }
  }
  rv[threadIdx.x] = bv; ri[threadIdx.x] = bi;
  __syncthreads();
  for (int s = 128; s > 0; s >>= 1) {
    if (threadIdx.x < (unsigned)s) {
      const float ov = rv[threadIdx.x + s]; const int oi = ri[threadIdx.x + s];
      if (ov > rv[threadIdx.x] || (ov == rv[threadIdx.x] && oi < ri[threadIdx.x])) {
        rv[threadIdx.x] = ov; ri[threadIdx.x] = oi;
      }
    }
    __syncthreads();
  }
  if (threadIdx.x == 0) out[blockIdx.x] = ri[0];
}

// ------------------------------------------------------------------
// K8: build Rp and Rn. One block per (q,b,f); float4 streaming copies.
// ------------------------------------------------------------------
__global__ __launch_bounds__(256) void build_outputs_kernel(const float* __restrict__ input,
                                                            const float* __restrict__ rscale,
                                                            const float* __restrict__ mem1,
                                                            const float* __restrict__ mem2,
                                                            const int* __restrict__ high_idx,
                                                            const int* __restrict__ low_idx,
                                                            const int* __restrict__ sim1,
                                                            const int* __restrict__ sim2,
                                                            float* __restrict__ Rp,
                                                            float* __restrict__ Rn) {
  const int blk = blockIdx.x;          // q*80 + b*20 + f
  const int f  = blk % F_;
  const int qb = blk / F_;             // q*4 + b
  const int b  = qb & 3;
  const int q  = qb >> 2;
  const int t  = b * Q_ + q;
  const int lo = low_idx[t];
  const int hi = high_idx[t];

  const float4* in4 = (const float4*)(input + (size_t)(b * F_ + f) * ROWB);
  const float*  scp = rscale + (b * F_ + f) * OBJ_;
  const bool useM1 = (f == lo);
  const bool useM2 = (f == hi);
  const float4* m1 = (const float4*)(mem1 + (size_t)sim1[b * F_ + f] * ROWB);
  const float4* m2 = (const float4*)(mem2 + (size_t)sim2[b * F_ + lo] * ROWB);
  float4* rp = (float4*)(Rp + (size_t)blk * ROWB);
  float4* rn = (float4*)(Rn + (size_t)blk * ROWB);

  for (int i = threadIdx.x; i < ROWB / 4; i += 256) {
    const int obj = i >> 10;                 // 1024 float4 per region
    const float s = scp[obj];
    const float4 v = in4[i];
    const float4 nv = make_float4(v.x * s, v.y * s, v.z * s, v.w * s);
    rp[i] = useM1 ? m1[i] : nv;
    rn[i] = useM2 ? m2[i] : nv;
  }
}

// K9: write high_sim / low_sim as floats into the output tail
__global__ void sims_out_kernel(const int* __restrict__ hsim,
                                const int* __restrict__ lsim,
                                float* __restrict__ out_tail) {
  const int i = threadIdx.x;
  if (i < 16)      out_tail[i] = (float)hsim[i];
  else if (i < 32) out_tail[i] = (float)lsim[i - 16];
}

// ------------------------------------------------------------------
extern "C" void kernel_launch(void* const* d_in, const int* in_sizes, int n_in,
                              void* d_out, int out_size, void* d_ws, size_t ws_size,
                              hipStream_t stream) {
  (void)in_sizes; (void)n_in; (void)out_size; (void)ws_size;
  const float* qr    = (const float*)d_in[0];
  const float* input = (const float*)d_in[1];
  const float* mem1  = (const float*)d_in[2];
  const float* mem2  = (const float*)d_in[3];
  const int*   rnd   = (const int*)d_in[4];

  _Float16* H = (_Float16*)d_ws;
  _Float16* mem1mp = H + H_MEM1MP;
  _Float16* mem2mp = H + H_MEM2MP;
  _Float16* rmax   = H + H_RMAX;
  _Float16* rhlmp  = H + H_RHLMP;

  float* Freg = (float*)((char*)d_ws + H_TOTAL * sizeof(_Float16));
  float* S1     = Freg + FOFF_S1;
  float* S2     = Freg + FOFF_S2;
  float* Sh     = Freg + FOFF_SH;
  float* Sl     = Freg + FOFF_SL;
  float* rscale = Freg + FOFF_RSCALE;
  int*   I      = (int*)(Freg + FOFF_INTS);
  int* high_idx = I + IHIGH;
  int* low_idx  = I + ILOW;
  int* rowmap   = I + IMAP;
  int* sim1     = I + ISIM1;
  int* sim2     = I + ISIM2;
  int* hsim     = I + IHSIM;
  int* lsim     = I + ILSIM;

  float* Rp = (float*)d_out;
  float* Rn = Rp + RP_ELEMS;
  float* out_tail = Rp + 2 * RP_ELEMS;

  // zero padding rows so every WMMA tile is fully in-bounds
  zero_pad_kernel<<<16, 256, 0, stream>>>(mem1mp, mem2mp);

  // memory-bank max-pool + normalize -> f16 (bandwidth-dominant: ~655 MB read)
  memmp_kernel<<<M_, 256, 0, stream>>>(mem1, mem1mp);
  memmp_kernel<<<M_, 256, 0, stream>>>(mem2, mem2mp);

  // per-region L2 scales of input features (read 26 MB once)
  rscale_kernel<<<B_ * F_ * OBJ_, 256, 0, stream>>>(input, rscale);

  // R_max for all B*F frames -> f16
  maxpool_norm_kernel<<<B_ * F_, 256, 0, stream>>>(input, rscale, nullptr, rmax);

  // frame selection + gather rows for high/low branches -> f16
  select_kernel<<<1, 32, 0, stream>>>(qr, rnd, high_idx, low_idx);
  rowmap_kernel<<<1, 32, 0, stream>>>(high_idx, low_idx, rowmap);
  maxpool_norm_kernel<<<32, 256, 0, stream>>>(input, rscale, rowmap, rhlmp);

  // WMMA similarity GEMMs + argmax (nearest memory slot)
  wmma_gemm_nt_kernel<<<5 * NT_N, 32, 0, stream>>>(rmax, mem1mp, S1, NT_N);
  wmma_gemm_nt_kernel<<<5 * NT_N, 32, 0, stream>>>(rmax, mem2mp, S2, NT_N);
  wmma_gemm_nt_kernel<<<1 * NT_N, 32, 0, stream>>>(rhlmp, mem2mp, Sh, NT_N);            // high vs mem2
  wmma_gemm_nt_kernel<<<1 * NT_N, 32, 0, stream>>>(rhlmp + 16 * D_, mem1mp, Sl, NT_N);  // low vs mem1
  argmax_kernel<<<80, 256, 0, stream>>>(S1, M_, sim1);
  argmax_kernel<<<80, 256, 0, stream>>>(S2, M_, sim2);
  argmax_kernel<<<16, 256, 0, stream>>>(Sh, M_, hsim);
  argmax_kernel<<<16, 256, 0, stream>>>(Sl, M_, lsim);

  // build Rp / Rn (writes 210 MB) and scalar sim outputs
  build_outputs_kernel<<<Q_ * B_ * F_, 256, 0, stream>>>(input, rscale, mem1, mem2,
                                                         high_idx, low_idx, sim1, sim2,
                                                         Rp, Rn);
  sims_out_kernel<<<1, 32, 0, stream>>>(hsim, lsim, out_tail);
}
